// MoELayer_52888227283711
// MI455X (gfx1250) — matrix operand order repaired and checked by reading, not compiled, not run on hardware
//
#include <hip/hip_runtime.h>
#include <hip/hip_bf16.h>
#include <stdint.h>

#define T_TOK 4096
#define DIM   512
#define NEXP  8
#define DFF   2048
#define MTILE 64
#define NC    512
#define XPITCH 520   // 512 + 8 halves: LDS row pad (16B aligned, rotates banks)

typedef unsigned short u16;
typedef __attribute__((ext_vector_type(16))) __bf16 v16bf;
typedef __attribute__((ext_vector_type(8)))  float  v8f;

union BFrag { v16bf v; uint4 q[2]; };

__device__ __forceinline__ u16 f2bf(float f) {
  unsigned u = __builtin_bit_cast(unsigned, f);
  unsigned r = (u + 0x7FFFu + ((u >> 16) & 1u)) >> 16;  // RNE
  return (u16)r;
}

__device__ __forceinline__ void atomic_add_f32(float* p, float v) {
  asm volatile("global_atomic_add_f32 %0, %1, off" :: "v"(p), "v"(v) : "memory");
}

// A-fragment (16x32 bf16, row-major source): lane L -> row (L&15);
// halves j=0..7 at K = 8*(L>>4)+j, j=8..15 at K = 16+8*(L>>4)+(j-8).
__device__ __forceinline__ v16bf load_frag_a(const u16* base /*row ptr at k0*/, int half) {
  BFrag f;
  f.q[0] = *(const uint4*)(base + 8 * half);
  f.q[1] = *(const uint4*)(base + 16 + 8 * half);
  return f.v;
}
// B-fragment (32x16 bf16) from K-contiguous (transposed) weights: lane L -> col (L&15),
// halves j=0..7 at K = 8*(L>>4)+j, j=8..15 at K = 16+8*(L>>4)+(j-8).
__device__ __forceinline__ v16bf load_frag_b(const u16* base /*row ptr at k0*/, int half) {
  BFrag f;
  f.q[0] = *(const uint4*)(base + 8 * half);
  f.q[1] = *(const uint4*)(base + 16 + 8 * half);
  return f.v;
}

__global__ void k_init(float* probSum, int* counts) {
  int t = threadIdx.x;
  if (t < NEXP) { probSum[t] = 0.f; counts[t] = 0; }
}

__global__ void k_convert_x(const float* __restrict__ x, u16* __restrict__ xb) {
  int i = blockIdx.x * blockDim.x + threadIdx.x;     // 8 elements per thread
  const float4* p = (const float4*)x + (size_t)i * 2;
  float4 a = p[0], b = p[1];
  uint4 r;
  r.x = (unsigned)f2bf(a.x) | ((unsigned)f2bf(a.y) << 16);
  r.y = (unsigned)f2bf(a.z) | ((unsigned)f2bf(a.w) << 16);
  r.z = (unsigned)f2bf(b.x) | ((unsigned)f2bf(b.y) << 16);
  r.w = (unsigned)f2bf(b.z) | ((unsigned)f2bf(b.w) << 16);
  *(uint4*)(xb + (size_t)i * 8) = r;
}

// src[e][R][C] fp32  ->  dst[e][C][R] bf16   (per-expert transpose+convert)
__global__ void k_transpose_bf16(const float* __restrict__ src, u16* __restrict__ dst,
                                 int R, int C) {
  __shared__ float tile[32][33];
  int tilesC = C >> 5;
  int tilesPerE = (R >> 5) * tilesC;
  int bx = blockIdx.x;
  int e = bx / tilesPerE;
  int rem = bx - e * tilesPerE;
  int tr = rem / tilesC;
  int tc = rem - tr * tilesC;
  int r0 = tr << 5, c0 = tc << 5;
  const float* s = src + (size_t)e * R * C;
  u16* d = dst + (size_t)e * R * C;
#pragma unroll
  for (int i = 0; i < 4; ++i)
    tile[threadIdx.y + 8 * i][threadIdx.x] =
        s[(size_t)(r0 + threadIdx.y + 8 * i) * C + c0 + threadIdx.x];
  __syncthreads();
#pragma unroll
  for (int i = 0; i < 4; ++i)
    d[(size_t)(c0 + threadIdx.y + 8 * i) * R + r0 + threadIdx.x] =
        f2bf(tile[threadIdx.x][threadIdx.y + 8 * i]);
}

// One wave per token: logits, softmax, argmax, lists, lb-loss partials.
__global__ void k_router(const float* __restrict__ x, const float* __restrict__ wr,
                         const float* __restrict__ br, float* __restrict__ out_logits,
                         float* __restrict__ tw, int* __restrict__ topidx,
                         int* __restrict__ counts, float* __restrict__ probSum,
                         int* __restrict__ lists) {
  int wv = threadIdx.x >> 5, lane = threadIdx.x & 31;
  int t = blockIdx.x * 8 + wv;
  float acc[NEXP];
#pragma unroll
  for (int e = 0; e < NEXP; ++e) acc[e] = 0.f;
  const float* xr = x + (size_t)t * DIM;
  for (int d = lane; d < DIM; d += 32) {
    float xv = xr[d];
    const float* w = wr + (size_t)d * NEXP;
#pragma unroll
    for (int e = 0; e < NEXP; ++e) acc[e] += xv * w[e];
  }
#pragma unroll
  for (int off = 16; off > 0; off >>= 1) {
#pragma unroll
    for (int e = 0; e < NEXP; ++e) acc[e] += __shfl_down(acc[e], off, 32);
  }
  if (lane == 0) {
    float lg[NEXP];
    float mx = -1e30f; int mi = 0;
#pragma unroll
    for (int e = 0; e < NEXP; ++e) {
      lg[e] = acc[e] + br[e];
      out_logits[(size_t)t * NEXP + e] = lg[e];
      if (lg[e] > mx) { mx = lg[e]; mi = e; }
    }
    float s = 0.f;
#pragma unroll
    for (int e = 0; e < NEXP; ++e) { lg[e] = __expf(lg[e] - mx); s += lg[e]; }
    float inv = 1.f / s;
#pragma unroll
    for (int e = 0; e < NEXP; ++e) atomic_add_f32(&probSum[e], lg[e] * inv);
    tw[t] = lg[mi] * inv;
    topidx[t] = mi;
    int pos = atomicAdd(&counts[mi], 1);
    lists[mi * T_TOK + pos] = t;
  }
}

// out[t][:] = top_w[t] * b2[expert_t][:]
__global__ void k_out_init(const float* __restrict__ b2, const float* __restrict__ tw,
                           const int* __restrict__ topidx, float* __restrict__ outy) {
  int t = blockIdx.x;
  int e = topidx[t];
  float w = tw[t];
  const float4* b = (const float4*)(b2 + (size_t)e * DIM);
  float4* o = (float4*)(outy + (size_t)t * DIM);
  float4 v = b[threadIdx.x];
  o[threadIdx.x] = make_float4(w * v.x, w * v.y, w * v.z, w * v.w);
}

__global__ void k_loss(const int* __restrict__ counts, const float* __restrict__ probSum,
                       float* __restrict__ out_lb) {
  if (threadIdx.x == 0) {
    float s = 0.f;
    for (int e = 0; e < NEXP; ++e)
      s += ((float)counts[e] / (float)T_TOK) * (probSum[e] / (float)T_TOK);
    *out_lb = (float)NEXP * s;
  }
}

// Fused per-expert FFN: block = (expert, 64-token tile), 8 waves.
// Each wave processes 2 N-groups x 4 M-tiles at once: every A-fragment
// (ds_load_b128 pair) feeds 2 WMMAs, every B-fragment feeds 4 WMMAs.
__global__ void __launch_bounds__(256) k_ffn(
    const u16* __restrict__ xb, const u16* __restrict__ w1t,
    const u16* __restrict__ w2t, const float* __restrict__ b1,
    const int* __restrict__ counts, const int* __restrict__ lists,
    const float* __restrict__ tw, float* __restrict__ outy) {
  int e = blockIdx.x >> 6;       // 64 worst-case tiles per expert
  int tix = blockIdx.x & 63;
  int cnt = counts[e];
  if (tix * MTILE >= cnt) return;

  extern __shared__ __align__(16) char smem[];
  u16* sx = (u16*)smem;                                   // 64 x XPITCH bf16
  u16* sh = (u16*)(smem + MTILE * XPITCH * 2);            // 64 x XPITCH bf16
  int* slst = (int*)(smem + 2 * MTILE * XPITCH * 2);
  float* stw = (float*)(smem + 2 * MTILE * XPITCH * 2 + 256);

  int tid = threadIdx.x;
  int wv = tid >> 5, lane = tid & 31;
  int half = lane >> 4, l15 = lane & 15;

  if (tid < MTILE) {
    int gm = tix * MTILE + tid;
    int tok = (gm < cnt) ? lists[e * T_TOK + gm] : 0;
    slst[tid] = tok;
    stw[tid] = (gm < cnt) ? tw[tok] : 0.f;   // pad rows contribute exactly 0
  }
  __syncthreads();

  // async gather of 64 bf16 token rows into LDS (16B per async op)
  for (int idx = tid; idx < MTILE * (DIM / 16); idx += 256) {
    int row = idx >> 5;            // 32 chunks per row
    int ch = idx & 31;
    unsigned ldsa = (unsigned)(size_t)(sx + row * XPITCH + ch * 16);
    const u16* ga = xb + (size_t)slst[row] * DIM + ch * 16;
    asm volatile("global_load_async_to_lds_b128 %0, %1, off"
                 :: "v"(ldsa), "v"(ga) : "memory");
  }
  asm volatile("s_wait_asynccnt 0x0" ::: "memory");
  __syncthreads();

  const u16* w1e = w1t + (size_t)e * DFF * DIM;   // [dff][d]  K-contiguous
  const u16* w2e = w2t + (size_t)e * DIM * DFF;   // [d][dff]  K-contiguous
  const float* b1e = b1 + (size_t)e * DFF;

  for (int c = 0; c < DFF / NC; ++c) {
    // ---- GEMM1: sh = top_w * relu(x @ W1[:,chunk] + b1) ----
#pragma unroll 1
    for (int g = 0; g < 2; ++g) {
      int ncloc[2];
      const u16* wrow[2];
      float bias[2];
#pragma unroll
      for (int p = 0; p < 2; ++p) {
        int nt = wv + 16 * g + 8 * p;              // disjoint per wave
        ncloc[p] = nt * 16 + l15;
        int ncol = c * NC + ncloc[p];              // this lane's dff column
        wrow[p] = w1e + (size_t)ncol * DIM;
        bias[p] = b1e[ncol];
      }
      v8f acc[2][4];
#pragma unroll
      for (int p = 0; p < 2; ++p)
#pragma unroll
        for (int mt = 0; mt < 4; ++mt)
#pragma unroll
          for (int r = 0; r < 8; ++r) acc[p][mt][r] = 0.f;
#pragma unroll 2
      for (int k0 = 0; k0 < DIM; k0 += 32) {
        v16bf bf0 = load_frag_b(wrow[0] + k0, half);
        v16bf bf1 = load_frag_b(wrow[1] + k0, half);
        v16bf a0 = load_frag_a(sx + (l15) * XPITCH + k0, half);
        v16bf a1 = load_frag_a(sx + (16 + l15) * XPITCH + k0, half);
        v16bf a2 = load_frag_a(sx + (32 + l15) * XPITCH + k0, half);
        v16bf a3 = load_frag_a(sx + (48 + l15) * XPITCH + k0, half);
        acc[0][0] = __builtin_amdgcn_wmma_f32_16x16x32_bf16(false, a0, false, bf0, (short)0, acc[0][0], false, false);
        acc[0][1] = __builtin_amdgcn_wmma_f32_16x16x32_bf16(false, a1, false, bf0, (short)0, acc[0][1], false, false);
        acc[0][2] = __builtin_amdgcn_wmma_f32_16x16x32_bf16(false, a2, false, bf0, (short)0, acc[0][2], false, false);
        acc[0][3] = __builtin_amdgcn_wmma_f32_16x16x32_bf16(false, a3, false, bf0, (short)0, acc[0][3], false, false);
        acc[1][0] = __builtin_amdgcn_wmma_f32_16x16x32_bf16(false, a0, false, bf1, (short)0, acc[1][0], false, false);
        acc[1][1] = __builtin_amdgcn_wmma_f32_16x16x32_bf16(false, a1, false, bf1, (short)0, acc[1][1], false, false);
        acc[1][2] = __builtin_amdgcn_wmma_f32_16x16x32_bf16(false, a2, false, bf1, (short)0, acc[1][2], false, false);
        acc[1][3] = __builtin_amdgcn_wmma_f32_16x16x32_bf16(false, a3, false, bf1, (short)0, acc[1][3], false, false);
      }
#pragma unroll
      for (int p = 0; p < 2; ++p)
#pragma unroll
        for (int mt = 0; mt < 4; ++mt) {
          const float4* wp = (const float4*)(stw + mt * 16 + 8 * half);
          float4 wa = wp[0], wb = wp[1];
          float wreg[8] = {wa.x, wa.y, wa.z, wa.w, wb.x, wb.y, wb.z, wb.w};
#pragma unroll
          for (int r = 0; r < 8; ++r) {
            int m = mt * 16 + 8 * half + r;
            float hv = fmaxf(acc[p][mt][r] + bias[p], 0.f) * wreg[r];  // branchless
            sh[m * XPITCH + ncloc[p]] = f2bf(hv);
          }
        }
    }
    __syncthreads();
    // ---- GEMM2: out += h_chunk @ W2[chunk,:] ----
#pragma unroll 1
    for (int g = 0; g < 2; ++g) {
      int dcol[2];
      const u16* wrow[2];
#pragma unroll
      for (int p = 0; p < 2; ++p) {
        int nt = wv + 16 * g + 8 * p;
        dcol[p] = nt * 16 + l15;
        wrow[p] = w2e + (size_t)dcol[p] * DFF + c * NC;
      }
      v8f acc[2][4];
#pragma unroll
      for (int p = 0; p < 2; ++p)
#pragma unroll
        for (int mt = 0; mt < 4; ++mt)
#pragma unroll
          for (int r = 0; r < 8; ++r) acc[p][mt][r] = 0.f;
#pragma unroll 2
      for (int k0 = 0; k0 < NC; k0 += 32) {
        v16bf bf0 = load_frag_b(wrow[0] + k0, half);
        v16bf bf1 = load_frag_b(wrow[1] + k0, half);
        v16bf a0 = load_frag_a(sh + (l15) * XPITCH + k0, half);
        v16bf a1 = load_frag_a(sh + (16 + l15) * XPITCH + k0, half);
        v16bf a2 = load_frag_a(sh + (32 + l15) * XPITCH + k0, half);
        v16bf a3 = load_frag_a(sh + (48 + l15) * XPITCH + k0, half);
        acc[0][0] = __builtin_amdgcn_wmma_f32_16x16x32_bf16(false, a0, false, bf0, (short)0, acc[0][0], false, false);
        acc[0][1] = __builtin_amdgcn_wmma_f32_16x16x32_bf16(false, a1, false, bf0, (short)0, acc[0][1], false, false);
        acc[0][2] = __builtin_amdgcn_wmma_f32_16x16x32_bf16(false, a2, false, bf0, (short)0, acc[0][2], false, false);
        acc[0][3] = __builtin_amdgcn_wmma_f32_16x16x32_bf16(false, a3, false, bf0, (short)0, acc[0][3], false, false);
        acc[1][0] = __builtin_amdgcn_wmma_f32_16x16x32_bf16(false, a0, false, bf1, (short)0, acc[1][0], false, false);
        acc[1][1] = __builtin_amdgcn_wmma_f32_16x16x32_bf16(false, a1, false, bf1, (short)0, acc[1][1], false, false);
        acc[1][2] = __builtin_amdgcn_wmma_f32_16x16x32_bf16(false, a2, false, bf1, (short)0, acc[1][2], false, false);
        acc[1][3] = __builtin_amdgcn_wmma_f32_16x16x32_bf16(false, a3, false, bf1, (short)0, acc[1][3], false, false);
      }
#pragma unroll
      for (int p = 0; p < 2; ++p)
#pragma unroll
        for (int mt = 0; mt < 4; ++mt) {
          const int4* tp = (const int4*)(slst + mt * 16 + 8 * half);
          int4 ta = tp[0], tb = tp[1];
          int toks[8] = {ta.x, ta.y, ta.z, ta.w, tb.x, tb.y, tb.z, tb.w};
#pragma unroll
          for (int r = 0; r < 8; ++r)
            atomic_add_f32(&outy[(size_t)toks[r] * DIM + dcol[p]], acc[p][mt][r]);
        }
    }
    __syncthreads();
  }
}

extern "C" void kernel_launch(void* const* d_in, const int* in_sizes, int n_in,
                              void* d_out, int out_size, void* d_ws, size_t ws_size,
                              hipStream_t stream) {
  (void)in_sizes; (void)n_in; (void)out_size; (void)ws_size;
  const float* x  = (const float*)d_in[0];
  const float* wr = (const float*)d_in[1];
  const float* br = (const float*)d_in[2];
  const float* W1 = (const float*)d_in[3];
  const float* b1 = (const float*)d_in[4];
  const float* W2 = (const float*)d_in[5];
  const float* b2 = (const float*)d_in[6];

  float* out = (float*)d_out;
  float* outy = out;                                  // [T][D]
  float* out_lb = out + (size_t)T_TOK * DIM;          // scalar
  float* out_logits = out_lb + 1;                     // [T][E]

  char* ws = (char*)d_ws;
  const size_t wbytes = (size_t)NEXP * DFF * DIM * 2; // 16.78 MB each
  u16* w1t = (u16*)ws;
  u16* w2t = (u16*)(ws + wbytes);
  u16* xb  = (u16*)(ws + 2 * wbytes);
  char* p  = ws + 2 * wbytes + (size_t)T_TOK * DIM * 2;
  float* tw   = (float*)p;  p += (size_t)T_TOK * 4;
  int* topidx = (int*)p;    p += (size_t)T_TOK * 4;
  int* lists  = (int*)p;    p += (size_t)NEXP * T_TOK * 4;
  int* counts = (int*)p;    p += 256;
  float* probSum = (float*)p;

  k_init<<<1, 32, 0, stream>>>(probSum, counts);
  k_convert_x<<<(T_TOK * DIM / 8) / 256, 256, 0, stream>>>(x, xb);
  k_transpose_bf16<<<NEXP * (DIM / 32) * (DFF / 32), dim3(32, 8), 0, stream>>>(W1, w1t, DIM, DFF);
  k_transpose_bf16<<<NEXP * (DFF / 32) * (DIM / 32), dim3(32, 8), 0, stream>>>(W2, w2t, DFF, DIM);
  k_router<<<T_TOK / 8, 256, 0, stream>>>(x, wr, br, out_logits, tw, topidx, counts, probSum, lists);
  k_out_init<<<T_TOK, 128, 0, stream>>>(b2, tw, topidx, outy);
  size_t smem = (size_t)2 * MTILE * XPITCH * 2 + 512;
  k_ffn<<<NEXP * 64, 256, smem, stream>>>(xb, w1t, w2t, b1, counts, lists, tw, outy);
  k_loss<<<1, 1, 0, stream>>>(counts, probSum, out_lb);
}
